// Encoder_62105227100798
// MI455X (gfx1250) — compile-verified, matrix operand order checked by reading, not run on hardware
//
#include <hip/hip_runtime.h>
#include <hip/hip_bf16.h>

// ---------------- problem constants ----------------
constexpr int kB  = 4;          // batch
constexpr int kS  = 2048;       // sequence
constexpr int kD  = 256;        // d_model
constexpr int kH  = 4;          // heads
constexpr int kDH = 64;         // depth per head
constexpr int kKV = 512;        // compressed kv length (S / MEM_COMP)
constexpr int kFF = 1024;       // dff
constexpr int kR  = kB * kS;    // total rows = 8192
constexpr int kIters = 8;       // MAX_ITER

// ---------------- CDNA5 async-to-LDS availability ----------------
#if defined(__AMDGCN__) && __has_builtin(__builtin_amdgcn_global_load_async_to_lds_b128) && \
    __has_builtin(__builtin_amdgcn_s_wait_asynccnt)
#define HAVE_ASYNC_LDS 1
typedef int v4i_async __attribute__((__vector_size__(16)));
#define GPTR(p) ((__attribute__((address_space(1))) v4i_async*)(p))
#define LPTR(p) ((__attribute__((address_space(3))) v4i_async*)(p))
#else
#define HAVE_ASYNC_LDS 0
#endif

// ---------------- WMMA types ----------------
typedef __attribute__((ext_vector_type(16))) __bf16 bf16x16;
typedef __attribute__((ext_vector_type(8)))  float  floatx8;

union Frag16 {
  uint4 q[2];
  bf16x16 v;
  unsigned short s[16];
};

__device__ __forceinline__ unsigned short f32_to_bf16(float f) {
  unsigned int u = __float_as_uint(f);
  u += 0x7FFFu + ((u >> 16) & 1u);   // round-to-nearest-even
  return (unsigned short)(u >> 16);
}

#define WMMA_BF16(Af, Bf, C) \
  __builtin_amdgcn_wmma_f32_16x16x32_bf16(false, (Af), false, (Bf), (short)0, (C), false, false)

// ---------------- utility kernels ----------------
__global__ void zero_f32_kernel(float* __restrict__ p, int n) {
  int i = blockIdx.x * blockDim.x + threadIdx.x;
  if (i < n) p[i] = 0.f;
}

// transpose-convert weights: in (K,N) f32 -> out (N,K) bf16. Dims % 16 == 0.
__global__ __launch_bounds__(256) void cvt_bf16_t_kernel(
    const float* __restrict__ in, unsigned short* __restrict__ out,
    int Kd, int Nd) {
  int kk = blockIdx.x * 16 + (threadIdx.x & 15);
  int nn = blockIdx.y * 16 + (threadIdx.x >> 4);
  out[(size_t)nn * Kd + kk] = f32_to_bf16(in[(size_t)kk * Nd + nn]);
}

// ---------------- pos-enc add + ACT halting ----------------
__global__ __launch_bounds__(256) void pos_act_kernel(
    float* __restrict__ state, const float* __restrict__ pos,
    const float* __restrict__ wp, const float* __restrict__ bp,
    const float* __restrict__ mask,
    float* __restrict__ halting, float* __restrict__ remainders,
    float* __restrict__ n_updates, float* __restrict__ uw,
    unsigned short* __restrict__ state_bf) {
  int wid = threadIdx.x >> 5, lane = threadIdx.x & 31;
  size_t row = (size_t)blockIdx.x * 8 + wid;          // row = b*S + s
  int s_in_b = (int)(row & (kS - 1));
  float* sp = state + row * kD;
  const float* pp = pos + (size_t)s_in_b * kD;
  float dot = 0.f;
  #pragma unroll
  for (int t = 0; t < 8; ++t) {
    int j = lane + 32 * t;
    float x = sp[j] + pp[j];
    sp[j] = x;
    state_bf[row * kD + j] = f32_to_bf16(x);
    dot += x * wp[j];
  }
  for (int o = 16; o; o >>= 1) dot += __shfl_xor(dot, o, 32);
  float p = 1.f / (1.f + __expf(-(dot + bp[0])));
  float hal = halting[row], rem = remainders[row], nu = n_updates[row];
  float m2 = mask[row];
  float still = (hal < 1.f) ? 1.f : 0.f;
  hal += m2;
  float t1 = hal + p * still;
  float nh  = (t1 >  0.99f) ? still : 0.f;
  float st2 = (t1 <= 0.99f) ? still : 0.f;
  hal += p * st2;
  rem += nh * (1.f - hal);
  hal += nh * rem;
  nu  += st2 + nh;
  float u = p * st2 + nh * rem;
  if (lane == 0) {
    halting[row] = hal; remainders[row] = rem; n_updates[row] = nu; uw[row] = u;
  }
}

// ---------------- bf16 tiled GEMM: C = act(A @ Bt^T + bias) ----------------
// A: (M,K) bf16 row-major, Bt: (N,K) bf16 row-major (pre-transposed weights).
// block = 256 threads = 8 waves; tile 128x128x32; 8 WMMAs/wave/K-step.
// Tiles staged via CDNA5 async-to-LDS DMA (double buffered) when available.
template <bool OUTF, bool OUTB, bool RELU>
__global__ __launch_bounds__(256) void gemm_bf16_kernel(
    const unsigned short* __restrict__ A, const unsigned short* __restrict__ Bt,
    const float* __restrict__ bias, float* __restrict__ outF,
    unsigned short* __restrict__ outB, int M, int N, int K) {
  __shared__ alignas(16) unsigned short As[2][128][40];
  __shared__ alignas(16) unsigned short Bs[2][128][40];   // [n][k]
  int tid = threadIdx.x;
  int wid = tid >> 5, lane = tid & 31;
  int nlane = lane & 15, hl = lane >> 4;
  int m0 = blockIdx.x * 128, n0 = blockIdx.y * 128;
  int row2 = tid >> 1, seg2 = tid & 1;   // 128 rows x (2 segs of 16 k)
  floatx8 acc[8] = {};

  const unsigned short* An = A  + (size_t)(m0 + row2) * K + seg2 * 16;
  const unsigned short* Bn = Bt + (size_t)(n0 + row2) * K + seg2 * 16;

  auto compute = [&](int bb) {
    Frag16 af;
    af.q[0] = *(const uint4*)&As[bb][wid * 16 + nlane][hl * 16];
    af.q[1] = *(const uint4*)&As[bb][wid * 16 + nlane][hl * 16 + 8];
    #pragma unroll
    for (int c = 0; c < 8; ++c) {
      Frag16 bfg;
      bfg.q[0] = *(const uint4*)&Bs[bb][c * 16 + nlane][hl * 16];
      bfg.q[1] = *(const uint4*)&Bs[bb][c * 16 + nlane][hl * 16 + 8];
      acc[c] = WMMA_BF16(af.v, bfg.v, acc[c]);
    }
  };

#if HAVE_ASYNC_LDS
  auto issue = [&](int k0, int bb) {
    __builtin_amdgcn_global_load_async_to_lds_b128(
        GPTR(An + k0),     LPTR(&As[bb][row2][seg2 * 16]),     0, 0);
    __builtin_amdgcn_global_load_async_to_lds_b128(
        GPTR(An + k0 + 8), LPTR(&As[bb][row2][seg2 * 16 + 8]), 0, 0);
    __builtin_amdgcn_global_load_async_to_lds_b128(
        GPTR(Bn + k0),     LPTR(&Bs[bb][row2][seg2 * 16]),     0, 0);
    __builtin_amdgcn_global_load_async_to_lds_b128(
        GPTR(Bn + k0 + 8), LPTR(&Bs[bb][row2][seg2 * 16 + 8]), 0, 0);
  };
  int buf = 0;
  issue(0, 0);
  int k0 = 0;
  for (; k0 + 32 < K; k0 += 32) {
    issue(k0 + 32, buf ^ 1);                 // stream next slice
    __builtin_amdgcn_s_wait_asynccnt(4);     // previous slice landed
    __syncthreads();
    compute(buf);
    __syncthreads();
    buf ^= 1;
  }
  __builtin_amdgcn_s_wait_asynccnt(0);
  __syncthreads();
  compute(buf);
#else
  uint4 a0 = *(const uint4*)(An);
  uint4 a1 = *(const uint4*)(An + 8);
  uint4 g0 = *(const uint4*)(Bn);
  uint4 g1 = *(const uint4*)(Bn + 8);
  for (int k0 = 0; k0 < K; k0 += 32) {
    *(uint4*)&As[0][row2][seg2 * 16]     = a0;
    *(uint4*)&As[0][row2][seg2 * 16 + 8] = a1;
    *(uint4*)&Bs[0][row2][seg2 * 16]     = g0;
    *(uint4*)&Bs[0][row2][seg2 * 16 + 8] = g1;
    __syncthreads();
    if (k0 + 32 < K) {
      a0 = *(const uint4*)(An + k0 + 32); a1 = *(const uint4*)(An + k0 + 40);
      g0 = *(const uint4*)(Bn + k0 + 32); g1 = *(const uint4*)(Bn + k0 + 40);
    }
    compute(0);
    __syncthreads();
  }
#endif

  #pragma unroll
  for (int c = 0; c < 8; ++c)
    #pragma unroll
    for (int r = 0; r < 8; ++r) {
      int row = m0 + wid * 16 + r + hl * 8;
      int col = n0 + c * 16 + nlane;
      float val = acc[c][r] + bias[col];
      if constexpr (RELU) val = fmaxf(val, 0.f);
      size_t idx = (size_t)row * N + col;
      if constexpr (OUTF) outF[idx] = val;
      if constexpr (OUTB) outB[idx] = f32_to_bf16(val);
    }
}

// ---------------- fused flash attention over compressed KV ----------------
// block = 8 waves on the SAME (b,h): 128 q rows; K tile async-DMA'd to LDS
// (double buffered), V tile register-staged + transposed into LDS.
__global__ __launch_bounds__(256) void attn_kernel(
    const unsigned short* __restrict__ qb,   // (B,S,256) bf16
    const unsigned short* __restrict__ kcb,  // (B,512,256) bf16
    const unsigned short* __restrict__ vcb,  // (B,512,256) bf16
    const float* __restrict__ mask,          // (B,S)
    unsigned short* __restrict__ ob) {       // (B,S,256) bf16
  __shared__ alignas(16) unsigned short Ks[2][32][72];  // [key][depth]
  __shared__ alignas(16) unsigned short Vs[64][40];     // [depth][key]
  __shared__ alignas(16) unsigned short Pl[8][16][32];  // per-wave P tile
  int tid = threadIdx.x;
  int wid = tid >> 5, lane = tid & 31;
  int nlane = lane & 15, hl = lane >> 4;
  int blk = blockIdx.x;              // kB*kH*(kS/128) = 256 blocks
  int qblk = blk & 15;
  int h = (blk >> 4) & 3;
  int b = blk >> 6;
  int q0 = qblk * 128 + wid * 16;
  int kt = tid >> 3, seg = tid & 7;  // loaders: 32 keys x 8 depth-segs of 8

  Frag16 qa0, qa1;
  {
    const unsigned short* qrow =
        qb + ((size_t)b * kS + q0 + nlane) * kD + h * kDH;
    qa0.q[0] = *(const uint4*)(qrow + hl * 16);
    qa0.q[1] = *(const uint4*)(qrow + hl * 16 + 8);
    qa1.q[0] = *(const uint4*)(qrow + 32 + hl * 16);
    qa1.q[1] = *(const uint4*)(qrow + 32 + hl * 16 + 8);
  }

  floatx8 oacc[4] = {};
  float rmax[8], rsum[8];
  #pragma unroll
  for (int r = 0; r < 8; ++r) { rmax[r] = -1e30f; rsum[r] = 0.f; }

  const unsigned short* kld = kcb + ((size_t)b * kKV + kt) * kD + h * kDH + seg * 8;
  const unsigned short* vld = vcb + ((size_t)b * kKV + kt) * kD + h * kDH + seg * 8;

  uint4 vv = *(const uint4*)(vld);
#if HAVE_ASYNC_LDS
  __builtin_amdgcn_global_load_async_to_lds_b128(
      GPTR(kld), LPTR(&Ks[0][kt][seg * 8]), 0, 0);
#else
  uint4 kv = *(const uint4*)(kld);
#endif
  int kbuf = 0;

  for (int g = 0; g < kKV / 32; ++g) {
    int kk = g * 32;
    __syncthreads();   // previous compute done; safe to refill Vs / other K buf
    union { uint4 q; unsigned short s[8]; } vu; vu.q = vv;
    #pragma unroll
    for (int j = 0; j < 8; ++j) Vs[seg * 8 + j][kt] = vu.s[j];
#if HAVE_ASYNC_LDS
    if (g + 1 < kKV / 32) {
      vv = *(const uint4*)(vld + (size_t)(kk + 32) * kD);
      __builtin_amdgcn_global_load_async_to_lds_b128(
          GPTR(kld + (size_t)(kk + 32) * kD),
          LPTR(&Ks[kbuf ^ 1][kt][seg * 8]), 0, 0);
      __builtin_amdgcn_s_wait_asynccnt(1);
    } else {
      __builtin_amdgcn_s_wait_asynccnt(0);
    }
#else
    *(uint4*)&Ks[0][kt][seg * 8] = kv;
    if (g + 1 < kKV / 32) {
      kv = *(const uint4*)(kld + (size_t)(kk + 32) * kD);
      vv = *(const uint4*)(vld + (size_t)(kk + 32) * kD);
    }
#endif
    __syncthreads();

    // S = Q K^T for 2 key chunks of 16
    floatx8 sacc[2] = {};
    #pragma unroll
    for (int c = 0; c < 2; ++c) {
      Frag16 kf;
      kf.q[0] = *(const uint4*)&Ks[kbuf][c * 16 + nlane][hl * 16];
      kf.q[1] = *(const uint4*)&Ks[kbuf][c * 16 + nlane][hl * 16 + 8];
      sacc[c] = WMMA_BF16(qa0.v, kf.v, sacc[c]);
      kf.q[0] = *(const uint4*)&Ks[kbuf][c * 16 + nlane][32 + hl * 16];
      kf.q[1] = *(const uint4*)&Ks[kbuf][c * 16 + nlane][32 + hl * 16 + 8];
      sacc[c] = WMMA_BF16(qa1.v, kf.v, sacc[c]);
    }
    float mk0 = mask[(size_t)b * kS + (kk + nlane) * 4]      * -1e9f;
    float mk1 = mask[(size_t)b * kS + (kk + 16 + nlane) * 4] * -1e9f;
    // online softmax (row stats per half-wave; reduce across 16 lanes)
    #pragma unroll
    for (int r = 0; r < 8; ++r) {
      float a0 = sacc[0][r] * 0.125f + mk0;
      float a1 = sacc[1][r] * 0.125f + mk1;
      float mx = fmaxf(a0, a1);
      for (int o = 8; o; o >>= 1) mx = fmaxf(mx, __shfl_xor(mx, o, 16));
      float nm = fmaxf(rmax[r], mx);
      float sc = __expf(rmax[r] - nm);
      rmax[r] = nm;
      float p0 = __expf(a0 - nm);
      float p1 = __expf(a1 - nm);
      float ps = p0 + p1;
      for (int o = 8; o; o >>= 1) ps += __shfl_xor(ps, o, 16);
      rsum[r] = rsum[r] * sc + ps;
      #pragma unroll
      for (int t = 0; t < 4; ++t) oacc[t][r] *= sc;
      int m = r + hl * 8;
      Pl[wid][m][nlane]      = f32_to_bf16(p0);
      Pl[wid][m][16 + nlane] = f32_to_bf16(p1);
    }
    // P (16x32) as A-fragment (same-wave LDS RAW; DS pipe in-order)
    Frag16 pf;
    pf.q[0] = *(const uint4*)&Pl[wid][nlane][hl * 16];
    pf.q[1] = *(const uint4*)&Pl[wid][nlane][hl * 16 + 8];
    // O += P @ V, 4 depth chunks of 16
    #pragma unroll
    for (int t = 0; t < 4; ++t) {
      Frag16 vf;
      vf.q[0] = *(const uint4*)&Vs[t * 16 + nlane][hl * 16];
      vf.q[1] = *(const uint4*)&Vs[t * 16 + nlane][hl * 16 + 8];
      oacc[t] = WMMA_BF16(pf.v, vf.v, oacc[t]);
    }
#if HAVE_ASYNC_LDS
    kbuf ^= 1;
#endif
  }
  #pragma unroll
  for (int t = 0; t < 4; ++t)
    #pragma unroll
    for (int r = 0; r < 8; ++r) {
      int m = r + hl * 8;
      float val = oacc[t][r] * (1.f / rsum[r]);
      ob[((size_t)b * kS + q0 + m) * kD + h * kDH + t * 16 + nlane] =
          f32_to_bf16(val);
    }
}

// ---------------- residual + LayerNorm (writes f32 + bf16) ----------------
__global__ __launch_bounds__(256) void ln_res_kernel(
    const float* __restrict__ a, const float* __restrict__ b,
    const float* __restrict__ g, const float* __restrict__ beta,
    float* __restrict__ outF, unsigned short* __restrict__ outB) {
  int wid = threadIdx.x >> 5, lane = threadIdx.x & 31;
  size_t row = (size_t)blockIdx.x * 8 + wid;
  const float* ap = a + row * kD;
  const float* bp = b + row * kD;
  float v[8]; float s = 0.f;
  #pragma unroll
  for (int t = 0; t < 8; ++t) { int j = lane + 32 * t; v[t] = ap[j] + bp[j]; s += v[t]; }
  for (int o = 16; o; o >>= 1) s += __shfl_xor(s, o, 32);
  float mu = s * (1.f / 256.f);
  float vs = 0.f;
  #pragma unroll
  for (int t = 0; t < 8; ++t) { float d = v[t] - mu; vs += d * d; }
  for (int o = 16; o; o >>= 1) vs += __shfl_xor(vs, o, 32);
  float inv = rsqrtf(vs * (1.f / 256.f) + 1e-6f);
  #pragma unroll
  for (int t = 0; t < 8; ++t) {
    int j = lane + 32 * t;
    float o1 = g[j] * (v[t] - mu) * inv + beta[j];
    outF[row * kD + j] = o1;
    outB[row * kD + j] = f32_to_bf16(o1);
  }
}

// ---------------- LN2 + ACT-weighted state update ----------------
__global__ __launch_bounds__(256) void ln_combine_kernel(
    const float* __restrict__ out1, const float* __restrict__ ffn,
    const float* __restrict__ g, const float* __restrict__ beta,
    const float* __restrict__ uw, float* __restrict__ state) {
  int wid = threadIdx.x >> 5, lane = threadIdx.x & 31;
  size_t row = (size_t)blockIdx.x * 8 + wid;
  const float* ap = out1 + row * kD;
  const float* bp = ffn + row * kD;
  float v[8]; float s = 0.f;
  #pragma unroll
  for (int t = 0; t < 8; ++t) { int j = lane + 32 * t; v[t] = ap[j] + bp[j]; s += v[t]; }
  for (int o = 16; o; o >>= 1) s += __shfl_xor(s, o, 32);
  float mu = s * (1.f / 256.f);
  float vs = 0.f;
  #pragma unroll
  for (int t = 0; t < 8; ++t) { float d = v[t] - mu; vs += d * d; }
  for (int o = 16; o; o >>= 1) vs += __shfl_xor(vs, o, 32);
  float inv = rsqrtf(vs * (1.f / 256.f) + 1e-6f);
  float u = uw[row];
  #pragma unroll
  for (int t = 0; t < 8; ++t) {
    int j = lane + 32 * t;
    float o2 = g[j] * (v[t] - mu) * inv + beta[j];
    float old = state[row * kD + j];
    state[row * kD + j] = o2 * u + old * (1.f - u);
  }
}

// ---------------- host orchestration ----------------
extern "C" void kernel_launch(void* const* d_in, const int* in_sizes, int n_in,
                              void* d_out, int out_size, void* d_ws, size_t ws_size,
                              hipStream_t stream) {
  (void)in_sizes; (void)n_in; (void)out_size; (void)ws_size;
  const float* x    = (const float*)d_in[0];
  const float* mask = (const float*)d_in[1];
  const float* pos  = (const float*)d_in[2];
  const float* wq = (const float*)d_in[3];  const float* bq = (const float*)d_in[4];
  const float* wk = (const float*)d_in[5];  const float* bk = (const float*)d_in[6];
  const float* wv = (const float*)d_in[7];  const float* bv = (const float*)d_in[8];
  const float* kcw = (const float*)d_in[9];  const float* kcb = (const float*)d_in[10];
  const float* vcw = (const float*)d_in[11]; const float* vcb = (const float*)d_in[12];
  const float* wo = (const float*)d_in[13]; const float* bo = (const float*)d_in[14];
  const float* w1 = (const float*)d_in[15]; const float* b1 = (const float*)d_in[16];
  const float* w2 = (const float*)d_in[17]; const float* b2 = (const float*)d_in[18];
  const float* ln1g = (const float*)d_in[19]; const float* ln1b = (const float*)d_in[20];
  const float* ln2g = (const float*)d_in[21]; const float* ln2b = (const float*)d_in[22];
  const float* wp = (const float*)d_in[23]; const float* bp = (const float*)d_in[24];

  float* state = (float*)d_out;                       // (R, 256) f32, final output

  char* ws = (char*)d_ws;
  size_t off = 0;
  auto alloc = [&](size_t bytes) -> void* {
    void* p = ws + off; off += (bytes + 255) & ~(size_t)255; return p;
  };
  // bf16 transposed weights (N,K)
  unsigned short* wq_b  = (unsigned short*)alloc((size_t)kD * kD * 2);
  unsigned short* wk_b  = (unsigned short*)alloc((size_t)kD * kD * 2);
  unsigned short* wv_b  = (unsigned short*)alloc((size_t)kD * kD * 2);
  unsigned short* kcw_b = (unsigned short*)alloc((size_t)4 * kD * kD * 2);
  unsigned short* vcw_b = (unsigned short*)alloc((size_t)4 * kD * kD * 2);
  unsigned short* wo_b  = (unsigned short*)alloc((size_t)kD * kD * 2);
  unsigned short* w1_b  = (unsigned short*)alloc((size_t)kD * kFF * 2);
  unsigned short* w2_b  = (unsigned short*)alloc((size_t)kFF * kD * 2);
  // activations
  unsigned short* state_bf = (unsigned short*)alloc((size_t)kR * kD * 2);
  unsigned short* q_bf  = (unsigned short*)alloc((size_t)kR * kD * 2);
  unsigned short* k_bf  = (unsigned short*)alloc((size_t)kR * kD * 2);  // reused as o_bf
  unsigned short* v_bf  = (unsigned short*)alloc((size_t)kR * kD * 2);  // reused as out1_bf
  unsigned short* kc_bf = (unsigned short*)alloc((size_t)kB * kKV * kD * 2);
  unsigned short* vc_bf = (unsigned short*)alloc((size_t)kB * kKV * kD * 2);
  unsigned short* h1_bf = (unsigned short*)alloc((size_t)kR * kFF * 2);
  float* scratchF = (float*)alloc((size_t)kR * kD * 4);  // attn_out, then ffn
  float* out1     = (float*)alloc((size_t)kR * kD * 4);
  float* actbuf   = (float*)alloc((size_t)3 * kR * 4);   // halting/remainders/n_updates
  float* uw       = (float*)alloc((size_t)kR * 4);
  float* halting = actbuf, *remainders = actbuf + kR, *n_updates = actbuf + 2 * kR;
  unsigned short* o_bf    = k_bf;  // k_full dead after compression GEMM
  unsigned short* out1_bf = v_bf;  // v_full dead after attention

  // init: state = x, ACT accumulators = 0, weights -> transposed bf16
  (void)hipMemcpyAsync(state, x, (size_t)kR * kD * sizeof(float),
                       hipMemcpyDeviceToDevice, stream);
  zero_f32_kernel<<<(3 * kR + 255) / 256, 256, 0, stream>>>(actbuf, 3 * kR);
  auto cvtT = [&](const float* src, unsigned short* dst, int Kd, int Nd) {
    dim3 g(Kd / 16, Nd / 16);
    cvt_bf16_t_kernel<<<g, 256, 0, stream>>>(src, dst, Kd, Nd);
  };
  cvtT(wq, wq_b, kD, kD);        cvtT(wk, wk_b, kD, kD);
  cvtT(wv, wv_b, kD, kD);        cvtT(kcw, kcw_b, 4 * kD, kD);
  cvtT(vcw, vcw_b, 4 * kD, kD);  cvtT(wo, wo_b, kD, kD);
  cvtT(w1, w1_b, kD, kFF);       cvtT(w2, w2_b, kFF, kD);

  auto gemm_bf = [&](const unsigned short* A, const unsigned short* Bt,
                     const float* bias, unsigned short* oB, int M, int N, int K) {
    dim3 grid(M / 128, N / 128);
    gemm_bf16_kernel<false, true, false><<<grid, 256, 0, stream>>>(
        A, Bt, bias, nullptr, oB, M, N, K);
  };
  auto gemm_f32 = [&](const unsigned short* A, const unsigned short* Bt,
                      const float* bias, float* oF, int M, int N, int K) {
    dim3 grid(M / 128, N / 128);
    gemm_bf16_kernel<true, false, false><<<grid, 256, 0, stream>>>(
        A, Bt, bias, oF, nullptr, M, N, K);
  };
  auto gemm_relu_bf = [&](const unsigned short* A, const unsigned short* Bt,
                          const float* bias, unsigned short* oB, int M, int N, int K) {
    dim3 grid(M / 128, N / 128);
    gemm_bf16_kernel<false, true, true><<<grid, 256, 0, stream>>>(
        A, Bt, bias, nullptr, oB, M, N, K);
  };

  for (int it = 0; it < kIters; ++it) {
    const float* pos_i = pos + (size_t)it * kS * kD;
    pos_act_kernel<<<kR / 8, 256, 0, stream>>>(state, pos_i, wp, bp, mask,
                                               halting, remainders, n_updates,
                                               uw, state_bf);
    // QKV projections
    gemm_bf(state_bf, wq_b, bq, q_bf, kR, kD, kD);
    gemm_bf(state_bf, wk_b, bk, k_bf, kR, kD, kD);
    gemm_bf(state_bf, wv_b, bv, v_bf, kR, kD, kD);
    // memory-compression convs == GEMM over flattened (window, channel)
    gemm_bf(k_bf, kcw_b, kcb, kc_bf, kB * kKV, kD, 4 * kD);
    gemm_bf(v_bf, vcw_b, vcb, vc_bf, kB * kKV, kD, 4 * kD);
    // fused flash attention
    attn_kernel<<<kB * kH * (kS / 128), 256, 0, stream>>>(
        q_bf, kc_bf, vc_bf, mask, o_bf);
    // output projection -> f32 residual branch
    gemm_f32(o_bf, wo_b, bo, scratchF, kR, kD, kD);
    ln_res_kernel<<<kR / 8, 256, 0, stream>>>(state, scratchF, ln1g, ln1b,
                                              out1, out1_bf);
    // FFN
    gemm_relu_bf(out1_bf, w1_b, b1, h1_bf, kR, kFF, kD);
    gemm_f32(h1_bf, w2_b, b2, scratchF, kR, kD, kFF);
    ln_combine_kernel<<<kR / 8, 256, 0, stream>>>(out1, scratchF, ln2g, ln2b,
                                                  uw, state);
  }
}
